// DynamicChunkingLayer_84619445666125
// MI455X (gfx1250) — matrix operand order repaired and checked by reading, not compiled, not run on hardware
//
#include <hip/hip_runtime.h>
#include <hip/hip_bf16.h>

// ---------------------------------------------------------------------------
// Problem constants (from reference): B=4, L=8192, D=1024
// ---------------------------------------------------------------------------
#define B_DIM 4
#define L_DIM 8192
#define D_DIM 1024
#define M_TOT (B_DIM * L_DIM)   // 32768 rows

// GEMM tiling
#define BM 128
#define BN 64
#define BK 32
#define LDSW 40                  // BK + 8 bf16 pad to dodge bank conflicts

typedef __attribute__((ext_vector_type(16))) __bf16 v16bf;
typedef __attribute__((ext_vector_type(8)))  float  v8f;

union FragBF {
    v16bf v;
    uint4 q2[2];
};

__device__ __forceinline__ unsigned short f2bf(float f) {
    // round-to-nearest-even float32 -> bf16 (NaN payload not preserved; fine here)
    unsigned int u = __float_as_uint(f);
    unsigned int r = (u + 0x7FFFu + ((u >> 16) & 1u)) >> 16;
    return (unsigned short)r;
}

// ---------------------------------------------------------------------------
// Kernel 1: fused Q/K GEMM.  Q = X * Wq^T, K = X * Wk^T  (bf16 WMMA, f32 acc)
// Block = 256 threads (8 wave32).  Block tile: 128(M) x 64(N), k-step 32.
// Wave w computes the 16x64 output strip for M-subtile w, for BOTH Q and K,
// reusing the A fragment (X read once per block).
// ---------------------------------------------------------------------------
__global__ __launch_bounds__(256) void gemm_qk_kernel(
    const float* __restrict__ x,
    const float* __restrict__ Wq,
    const float* __restrict__ Wk,
    float* __restrict__ qout,
    float* __restrict__ kout)
{
    __shared__ unsigned short As [BM][LDSW];
    __shared__ unsigned short Bqs[BN][LDSW];
    __shared__ unsigned short Bks[BN][LDSW];

    const int tid  = threadIdx.x;
    const int wave = tid >> 5;          // 0..7 (wave32)
    const int lane = tid & 31;
    const int m0   = blockIdx.x * BM;
    const int n0   = blockIdx.y * BN;

    v8f accQ[4], accK[4];
    for (int n = 0; n < 4; ++n)
        for (int e = 0; e < 8; ++e) { accQ[n][e] = 0.0f; accK[n][e] = 0.0f; }

    const int g = lane >> 4;            // K-half selector (contiguous k order)
    const int r = lane & 15;            // M (or N) index within 16x16 subtile

    for (int k0 = 0; k0 < D_DIM; k0 += BK) {
        // ---- stage A tile (128x32 f32 -> bf16): 1024 float4, 4 per thread ----
        for (int j = 0; j < 4; ++j) {
            int idx = tid + j * 256;            // 0..1023
            int row = idx >> 3;                 // 0..127
            int c4  = (idx & 7) << 2;           // 0,4,..,28
            const float4 f = *(const float4*)(&x[(size_t)(m0 + row) * D_DIM + k0 + c4]);
            As[row][c4 + 0] = f2bf(f.x);
            As[row][c4 + 1] = f2bf(f.y);
            As[row][c4 + 2] = f2bf(f.z);
            As[row][c4 + 3] = f2bf(f.w);
        }
        // ---- stage Bq/Bk tiles (64x32 each): 512 float4, 2 per thread each ----
        for (int j = 0; j < 2; ++j) {
            int idx = tid + j * 256;            // 0..511
            int row = idx >> 3;                 // 0..63
            int c4  = (idx & 7) << 2;
            const size_t off = (size_t)(n0 + row) * D_DIM + k0 + c4;
            const float4 fq = *(const float4*)(&Wq[off]);
            const float4 fk = *(const float4*)(&Wk[off]);
            Bqs[row][c4 + 0] = f2bf(fq.x); Bqs[row][c4 + 1] = f2bf(fq.y);
            Bqs[row][c4 + 2] = f2bf(fq.z); Bqs[row][c4 + 3] = f2bf(fq.w);
            Bks[row][c4 + 0] = f2bf(fk.x); Bks[row][c4 + 1] = f2bf(fk.y);
            Bks[row][c4 + 2] = f2bf(fk.z); Bks[row][c4 + 3] = f2bf(fk.w);
        }
        __syncthreads();

        // ---- gather fragments, same contiguous k order in A and B ----
        // A 16x32 bf16: lane holds M=r; K = g*16 .. g*16+15 (contiguous)
        FragBF a;
        a.q2[0] = *(const uint4*)(&As[wave * 16 + r][g * 16]);
        a.q2[1] = *(const uint4*)(&As[wave * 16 + r][g * 16 + 8]);

        FragBF bq[4], bk[4];
        for (int n = 0; n < 4; ++n) {
            // B 32x16 bf16: lane holds N=r; K = g*16 .. g*16+15 (contiguous)
            bq[n].q2[0] = *(const uint4*)(&Bqs[n * 16 + r][g * 16]);
            bq[n].q2[1] = *(const uint4*)(&Bqs[n * 16 + r][g * 16 + 8]);
            bk[n].q2[0] = *(const uint4*)(&Bks[n * 16 + r][g * 16]);
            bk[n].q2[1] = *(const uint4*)(&Bks[n * 16 + r][g * 16 + 8]);
        }

        for (int n = 0; n < 4; ++n) {
            accQ[n] = __builtin_amdgcn_wmma_f32_16x16x32_bf16(
                false, a.v, false, bq[n].v, (short)0, accQ[n], false, false);
            accK[n] = __builtin_amdgcn_wmma_f32_16x16x32_bf16(
                false, a.v, false, bk[n].v, (short)0, accK[n], false, false);
        }
        __syncthreads();
    }

    // ---- epilogue: D-matrix layout: VGPR v holds M = v + 8*g, N = lane&15 ----
    for (int n = 0; n < 4; ++n) {
        const int col = n0 + n * 16 + r;
        for (int v = 0; v < 8; ++v) {
            const int row = m0 + wave * 16 + v + 8 * g;
            qout[(size_t)row * D_DIM + col] = accQ[n][v];
            kout[(size_t)row * D_DIM + col] = accK[n][v];
        }
    }
}

// ---------------------------------------------------------------------------
// Kernel 2: per-row norms + shifted cosine -> p, b; atomic sums for F/G means
// ---------------------------------------------------------------------------
__global__ __launch_bounds__(256) void rownorm_kernel(
    const float* __restrict__ q, const float* __restrict__ k,
    float* __restrict__ p_out, float* __restrict__ b_out,
    float* __restrict__ sums)
{
    const int r   = blockIdx.x;          // 0 .. M_TOT-1
    const int l   = r & (L_DIM - 1);
    const int tid = threadIdx.x;

    float sq = 0.f, sk = 0.f, dp = 0.f;
    if (l > 0) {
        const float* qrow = q + (size_t)r * D_DIM;
        const float* krow = k + (size_t)(r - 1) * D_DIM;
        for (int i = tid; i < D_DIM; i += 256) {
            float qa = qrow[i], ka = krow[i];
            sq += qa * qa;
            sk += ka * ka;
            dp += qa * ka;
        }
    }
    __shared__ float s0[256], s1[256], s2[256];
    s0[tid] = sq; s1[tid] = sk; s2[tid] = dp;
    __syncthreads();
    for (int off = 128; off > 0; off >>= 1) {
        if (tid < off) {
            s0[tid] += s0[tid + off];
            s1[tid] += s1[tid + off];
            s2[tid] += s2[tid + off];
        }
        __syncthreads();
    }
    if (tid == 0) {
        float pval, bval;
        if (l == 0) {
            pval = 1.0f; bval = 1.0f;
        } else {
            float nq = fmaxf(sqrtf(s0[0]), 1e-12f);
            float nk = fmaxf(sqrtf(s1[0]), 1e-12f);
            float cs = s2[0] / (nq * nk);
            pval = 0.5f * (1.0f - cs);
            bval = (pval >= 0.5f) ? 1.0f : 0.0f;
        }
        p_out[r] = pval;
        b_out[r] = bval;
        atomicAdd(&sums[0], bval);
        atomicAdd(&sums[1], pval);
    }
}

// ---------------------------------------------------------------------------
// Kernel 3: per-batch inclusive scan of b -> dest slots + lengths
// ---------------------------------------------------------------------------
__global__ __launch_bounds__(256) void scan_kernel(
    const float* __restrict__ b_out, int* __restrict__ dest,
    float* __restrict__ len_out)
{
    const int batch = blockIdx.x;
    const int tid   = threadIdx.x;
    __shared__ int tmp[256];
    __shared__ int carry;
    if (tid == 0) carry = 0;
    __syncthreads();

    for (int base = 0; base < L_DIM; base += 256) {
        const int l = base + tid;
        int v = (b_out[batch * L_DIM + l] > 0.5f) ? 1 : 0;
        tmp[tid] = v;
        __syncthreads();
        for (int off = 1; off < 256; off <<= 1) {
            int t = (tid >= off) ? tmp[tid - off] : 0;
            __syncthreads();
            tmp[tid] += t;
            __syncthreads();
        }
        const int total = tmp[255];
        int d = carry + tmp[tid] - 1;
        d = min(max(d, 0), L_DIM - 1);
        dest[batch * L_DIM + l] = d;
        __syncthreads();
        if (tid == 0) carry += total;
        __syncthreads();
    }
    if (tid == 0) len_out[batch] = (float)carry;
}

// ---------------------------------------------------------------------------
// Kernel 4: masked row scatter (stream compaction)
// ---------------------------------------------------------------------------
__global__ __launch_bounds__(256) void scatter_kernel(
    const float* __restrict__ x, const float* __restrict__ b_out,
    const float* __restrict__ p_out, const int* __restrict__ dest,
    float* __restrict__ x_down, float* __restrict__ P_down)
{
    const int r = blockIdx.x;            // 0 .. M_TOT-1
    if (b_out[r] <= 0.5f) return;
    const int batch = r / L_DIM;
    const int d     = dest[r];
    const size_t src = (size_t)r * D_DIM;
    const size_t dst = ((size_t)batch * L_DIM + d) * D_DIM;
    for (int i = threadIdx.x; i < D_DIM / 4; i += 256) {
        float4 v = *(const float4*)(&x[src + 4 * i]);
        *(float4*)(&x_down[dst + 4 * i]) = v;
    }
    if (threadIdx.x == 0) P_down[batch * L_DIM + d] = p_out[r];
}

// ---------------------------------------------------------------------------
// Small helpers
// ---------------------------------------------------------------------------
__global__ void zero_kernel(float* ptr, size_t n) {
    size_t i = (size_t)blockIdx.x * blockDim.x + threadIdx.x;
    if (i < n) ptr[i] = 0.0f;
}

__global__ void finalize_kernel(const float* __restrict__ sums,
                                float* __restrict__ loss_out) {
    const float inv = 1.0f / (float)(M_TOT);
    float F = sums[0] * inv;
    float G = sums[1] * inv;
    const float N = 6.0f;
    loss_out[0] = N / (N - 1.0f) * ((N - 1.0f) * F * G + (1.0f - F) * (1.0f - G));
}

// ---------------------------------------------------------------------------
extern "C" void kernel_launch(void* const* d_in, const int* in_sizes, int n_in,
                              void* d_out, int out_size, void* d_ws, size_t ws_size,
                              hipStream_t stream) {
    (void)in_sizes; (void)n_in; (void)out_size; (void)ws_size;

    const float* x  = (const float*)d_in[0];
    const float* Wq = (const float*)d_in[1];
    const float* Wk = (const float*)d_in[2];

    // Output layout (reference tuple, concatenated flat):
    float* out      = (float*)d_out;
    float* x_down   = out;                                   // B*L*D
    float* P_down   = x_down + (size_t)M_TOT * D_DIM;        // B*L
    float* b_out    = P_down + M_TOT;                        // B*L
    float* p_out    = b_out + M_TOT;                         // B*L
    float* len_out  = p_out + M_TOT;                         // B
    float* loss_out = len_out + B_DIM;                       // 1

    // Workspace layout: q (f32), k (f32), dest (i32), sums (2 f32)
    float* qbuf = (float*)d_ws;
    float* kbuf = qbuf + (size_t)M_TOT * D_DIM;
    int*   dest = (int*)(kbuf + (size_t)M_TOT * D_DIM);
    float* sums = (float*)(dest + M_TOT);

    // 0) zero accumulators and the scattered outputs
    zero_kernel<<<1, 64, 0, stream>>>(sums, 2);
    {
        size_t nz = (size_t)M_TOT * D_DIM + M_TOT;           // x_down + P_down
        int blocks = (int)((nz + 255) / 256);
        zero_kernel<<<blocks, 256, 0, stream>>>(x_down, nz);
    }

    // 1) fused Q/K GEMM (bf16 WMMA, f32 accumulate)
    gemm_qk_kernel<<<dim3(M_TOT / BM, D_DIM / BN), 256, 0, stream>>>(
        x, Wq, Wk, qbuf, kbuf);

    // 2) norms + shifted cosine -> p, b (+ F/G partial sums)
    rownorm_kernel<<<M_TOT, 256, 0, stream>>>(qbuf, kbuf, p_out, b_out, sums);

    // 3) per-batch scan -> dest, lengths
    scan_kernel<<<B_DIM, 256, 0, stream>>>(b_out, dest, len_out);

    // 4) masked scatter -> x_down, P_down
    scatter_kernel<<<M_TOT, 256, 0, stream>>>(x, b_out, p_out, dest, x_down, P_down);

    // 5) scalar ratio loss
    finalize_kernel<<<1, 1, 0, stream>>>(sums, loss_out);
}
